// BDHAttention_89386859364510
// MI455X (gfx1250) — matrix operand order repaired and checked by reading, not compiled, not run on hardware
//
#include <hip/hip_runtime.h>
#include <math.h>

typedef __attribute__((ext_vector_type(16))) _Float16 v16h;
typedef __attribute__((ext_vector_type(8)))  _Float16 v8h;
typedef __attribute__((ext_vector_type(2)))  _Float16 v2h;
typedef __attribute__((ext_vector_type(8)))  float    v8f;
typedef __attribute__((ext_vector_type(2)))  float    v2f;

#define BH   8
#define TDIM 2048
#define NDIM 4096
#define RS   40   // padded LDS row stride in halves (80B -> conflict-free b128 frags)

#define TILE_BYTES (128 * RS * 2)      // 10240 B per tile
#define BUF_BYTES  (2 * TILE_BYTES)    // 20480 B per double-buffer slot

static __device__ inline void wait_async0() {
#if __has_builtin(__builtin_amdgcn_s_wait_asynccnt)
  __builtin_amdgcn_s_wait_asynccnt(0);
#else
  asm volatile("s_wait_asynccnt 0" ::: "memory");
#endif
}

// one 16B async global->LDS transfer per lane (ASYNCcnt-tracked)
static __device__ inline void async_copy16(unsigned lds_byte_off, const _Float16* gaddr) {
  asm volatile("global_load_async_to_lds_b128 %0, %1, off"
               :: "v"(lds_byte_off), "v"(gaddr)
               : "memory");
}

// ---------------- RoPE fp32 -> f16 (Q -> QRh, row-major [BH,T,N]) ----------------
__global__ void rope_kernel(const float* __restrict__ Q, _Float16* __restrict__ QRh) {
  const float inv2pi = 0.15915494309189535f;
  long p = (long)blockIdx.x * blockDim.x + threadIdx.x;  // pair index
  const int halfN = NDIM / 2;
  int  np  = (int)(p % halfN);
  long row = p / halfN;                                  // [0, BH*T)
  int  t   = (int)(row % TDIM);
  // freq = theta^(-floor(n/2)*2/N)/(2pi), theta=2^16 -> exp2(-(n/2)/128)/(2pi)
  float f  = exp2f(-(float)np * (1.0f / 128.0f)) * inv2pi;
  float ph = (float)t * f;
  ph = (ph - floorf(ph)) * 6.283185307179586f;
  float s, c;
  __sincosf(ph, &s, &c);
  v2f q = *(const v2f*)(Q + row * NDIM + 2 * np);
  v2h o;
  o.x = (_Float16)(q.x * c - q.y * s);
  o.y = (_Float16)(q.y * c + q.x * s);
  *(v2h*)(QRh + row * NDIM + 2 * np) = o;
}

// ---------------- V fp32 [BH,T,N] -> f16 transposed [BH,N,T] ----------------
__global__ void transpose_v_kernel(const float* __restrict__ V, _Float16* __restrict__ Vth) {
  __shared__ _Float16 tile[32][33];
  int bh = blockIdx.z;
  int n0 = blockIdx.x * 32;
  int t0 = blockIdx.y * 32;
  const float* Vb = V + (size_t)bh * TDIM * NDIM;
  _Float16*    Ob = Vth + (size_t)bh * NDIM * TDIM;
  int tx = threadIdx.x, ty = threadIdx.y;
#pragma unroll
  for (int i = 0; i < 4; ++i) {
    int r = ty + i * 8;
    tile[r][tx] = (_Float16)Vb[(size_t)(t0 + r) * NDIM + n0 + tx];
  }
  __syncthreads();
#pragma unroll
  for (int i = 0; i < 4; ++i) {
    int r = ty + i * 8;
    Ob[(size_t)(n0 + r) * TDIM + t0 + tx] = tile[tx][r];
  }
}

// ---------------- tiled f16 GEMM with fp32 WMMA accumulation ----------------
// C[row,col] = scale * sum_k A[row,k] * B[col,k]   (B given "col-major as rows")
// Block tile 128x128, 8 waves of 32 lanes, wave tile 32x64 (2x4 wmma frags), BK=32.
// Tile staging via GLOBAL_LOAD_ASYNC_TO_LDS_B128 (no VGPR round-trip), double-buffered.
template <bool F16OUT>
__global__ __launch_bounds__(256) void gemm_wmma_kernel(
    const _Float16* __restrict__ A, const _Float16* __restrict__ B,
    void* __restrict__ outp, int K, int lda, int ldb, int ldo,
    long aStride, long bStride, long oStride, float scale) {
  __shared__ _Float16 smem[2][2][128 * RS];  // [buf][A/B][rows*RS] = 40KB

  int bh = blockIdx.z;
  A += (size_t)bh * aStride;
  B += (size_t)bh * bStride;
  int row0 = blockIdx.y * 128;
  int col0 = blockIdx.x * 128;
  const _Float16* Ag = A + (size_t)row0 * lda;
  const _Float16* Bg = B + (size_t)col0 * ldb;

  int tid = threadIdx.x;
  int lane = tid & 31, w = tid >> 5;
  int wr = w & 3, wc = w >> 2;         // 4x2 wave grid
  int lrow = lane & 15, lhi = lane >> 4;

  v8f zero = {};
  v8f acc[2][4];
#pragma unroll
  for (int mi = 0; mi < 2; ++mi)
#pragma unroll
    for (int ni = 0; ni < 4; ++ni) acc[mi][ni] = zero;

  // per-thread staging segments: A tile 128x32 + B tile 128x32 halves, 4 x 16B each
  int srow[2], scol[2];
#pragma unroll
  for (int i = 0; i < 2; ++i) {
    int s = tid + i * 256;
    srow[i] = s >> 2;
    scol[i] = (s & 3) * 8;
  }
  // wave-relative LDS byte offset of smem base (flat-shared aperture keeps it in addr[31:0])
  unsigned lbase = (unsigned)(uintptr_t)&smem[0][0][0];
  unsigned offA[2], offB[2];
#pragma unroll
  for (int i = 0; i < 2; ++i) {
    offA[i] = lbase + (unsigned)(srow[i] * RS + scol[i]) * 2u;
    offB[i] = offA[i] + TILE_BYTES;
  }

  auto issue_async = [&](int kt, int buf) {
    int k0 = kt * 32;
    unsigned bo = (unsigned)buf * BUF_BYTES;
#pragma unroll
    for (int i = 0; i < 2; ++i) {
      async_copy16(offA[i] + bo, Ag + (size_t)srow[i] * lda + k0 + scol[i]);
      async_copy16(offB[i] + bo, Bg + (size_t)srow[i] * ldb + k0 + scol[i]);
    }
  };

  int KT = K >> 5;
  issue_async(0, 0);
  wait_async0();
  __syncthreads();

  for (int kt = 0; kt < KT; ++kt) {
    int buf = kt & 1;
    if (kt + 1 < KT) issue_async(kt + 1, buf ^ 1);  // DMA next tile, overlaps compute

    const _Float16* La = &smem[buf][0][0];
    const _Float16* Lb = &smem[buf][1][0];
    v16h aF[2], bF[4];
#pragma unroll
    for (int mi = 0; mi < 2; ++mi) {
      int r = wr * 32 + mi * 16 + lrow;
      v8h lo = *(const v8h*)&La[r * RS + lhi * 8];
      v8h hi = *(const v8h*)&La[r * RS + lhi * 8 + 16];
#pragma unroll
      for (int j = 0; j < 8; ++j) { aF[mi][j] = lo[j]; aF[mi][j + 8] = hi[j]; }
    }
#pragma unroll
    for (int ni = 0; ni < 4; ++ni) {
      int r = wc * 64 + ni * 16 + lrow;
      v8h lo = *(const v8h*)&Lb[r * RS + lhi * 8];
      v8h hi = *(const v8h*)&Lb[r * RS + lhi * 8 + 16];
#pragma unroll
      for (int j = 0; j < 8; ++j) { bF[ni][j] = lo[j]; bF[ni][j + 8] = hi[j]; }
    }
#pragma unroll
    for (int mi = 0; mi < 2; ++mi)
#pragma unroll
      for (int ni = 0; ni < 4; ++ni)
        acc[mi][ni] = __builtin_amdgcn_wmma_f32_16x16x32_f16(
            false, aF[mi], false, bF[ni], (short)0, acc[mi][ni], false, false);

    wait_async0();     // own async transfers for next tile are done
    __syncthreads();   // everyone done reading buf / writing buf^1
  }

  // epilogue: C layout M = v + 8*(lane/16), N = lane%16
#pragma unroll
  for (int mi = 0; mi < 2; ++mi)
#pragma unroll
    for (int ni = 0; ni < 4; ++ni) {
      int rb = row0 + wr * 32 + mi * 16 + lhi * 8;
      int cb = col0 + wc * 64 + ni * 16 + lrow;
      if constexpr (F16OUT) {
        _Float16* O = (_Float16*)outp + (size_t)bh * oStride;
#pragma unroll
        for (int v = 0; v < 8; ++v)
          O[(size_t)(rb + v) * ldo + cb] = (_Float16)(acc[mi][ni][v] * scale);
      } else {
        float* O = (float*)outp + (size_t)bh * oStride;
#pragma unroll
        for (int v = 0; v < 8; ++v)
          O[(size_t)(rb + v) * ldo + cb] = acc[mi][ni][v] * scale;
      }
    }
}

extern "C" void kernel_launch(void* const* d_in, const int* in_sizes, int n_in,
                              void* d_out, int out_size, void* d_ws, size_t ws_size,
                              hipStream_t stream) {
  (void)in_sizes; (void)n_in; (void)out_size; (void)ws_size;
  const float* Q = (const float*)d_in[0];
  // d_in[1] = K is unused by the reference (scores use QR twice)
  const float* V = (const float*)d_in[2];
  float* out = (float*)d_out;

  _Float16* QRh = (_Float16*)d_ws;                          // 128 MB
  _Float16* Vth = QRh + (size_t)BH * TDIM * NDIM;           // 128 MB
  _Float16* Sh  = Vth + (size_t)BH * TDIM * NDIM;           // 64 MB

  long pairs = (long)BH * TDIM * NDIM / 2;
  rope_kernel<<<(int)(pairs / 256), 256, 0, stream>>>(Q, QRh);
  transpose_v_kernel<<<dim3(NDIM / 32, TDIM / 32, BH), dim3(32, 8), 0, stream>>>(V, Vth);

  const float scale = 1.0f / 64.0f;  // 1/sqrt(N), N=4096
  // S = scale * QR . QR^T  (both tiles from row-major QR; S is symmetric)
  gemm_wmma_kernel<true><<<dim3(TDIM / 128, TDIM / 128, BH), 256, 0, stream>>>(
      QRh, QRh, Sh, NDIM, NDIM, NDIM, TDIM,
      (long)TDIM * NDIM, (long)TDIM * NDIM, (long)TDIM * TDIM, scale);
  // out = S . V   (B tiles from pre-transposed V)
  gemm_wmma_kernel<false><<<dim3(NDIM / 128, TDIM / 128, BH), 256, 0, stream>>>(
      Sh, Vth, out, TDIM, TDIM, TDIM, NDIM,
      (long)TDIM * TDIM, (long)NDIM * TDIM, (long)TDIM * NDIM, 1.0f);
}
